// ParticleFilter_32401233281657
// MI455X (gfx1250) — compile-verified, matrix operand order checked by reading, not run on hardware
//
#include <hip/hip_runtime.h>
#include <hip/hip_bf16.h>

#define NB   16     // batch
#define NT   64     // time steps
#define DS   64     // state dim
#define DOBS 32     // obs dim
#define NP   512    // particles
#define KA   160    // layer0 K = DS(noise)+DOBS(obs)+DS(h0)
#define KB   128    // layer1 K = DS(h0new)+DS(h1old)
#define GATES 256   // 4*DS
#define PANSZ (NP * DS)

typedef __attribute__((ext_vector_type(16))) __bf16 v16bf;
typedef __attribute__((ext_vector_type(8)))  float  v8f;

union BFrag { v16bf v; unsigned short s[16]; uint4 q[2]; };

// hardware f32->bf16 (lowers to v_cvt_*_bf16_f32, packable) instead of the
// 3-VALU integer RNE sequence
static __device__ __forceinline__ unsigned short f2bf(float f) {
  __bf16 h = (__bf16)f;
  unsigned short s;
  __builtin_memcpy(&s, &h, 2);
  return s;
}
static __device__ __forceinline__ float bf2f(unsigned short h) {
  union { unsigned u; float f; } x; x.u = ((unsigned)h) << 16;
  return x.f;
}
static __device__ __forceinline__ unsigned hashu(unsigned a) {
  a ^= a >> 16; a *= 0x7feb352du; a ^= a >> 15; a *= 0x846ca68bu; a ^= a >> 16;
  return a;
}
static __device__ __forceinline__ float u01(unsigned h) {
  return (float)(h >> 8) * (1.0f / 16777216.0f);
}
// Box-Muller pair: one hash chain + one log/sqrt + cos&sin -> two normals
static __device__ __forceinline__ void nrm2(unsigned seed, float* n1, float* n2) {
  unsigned h1 = hashu(seed * 0x9E3779B9u + 0x85EBCA6Bu);
  unsigned h2 = hashu(h1 ^ 0xC2B2AE35u);
  float r   = sqrtf(-2.0f * __logf(u01(h1) + 1e-7f));
  float ang = 6.2831853f * u01(h2);
  *n1 = r * __cosf(ang);
  *n2 = r * __sinf(ang);
}
static __device__ __forceinline__ float sigm(float x) { return 1.0f / (1.0f + __expf(-x)); }
static __device__ __forceinline__ float tanh_f(float x) {
  float e = __expf(2.0f * x); return (e - 1.0f) / (e + 1.0f);
}

static __device__ __forceinline__ v16bf fragZero() {
  BFrag u; u.q[0] = make_uint4(0,0,0,0); u.q[1] = make_uint4(0,0,0,0); return u.v;
}
// 16-bit A/B WMMA layout: per lane, VGPRs 0-3 hold K=base..base+7, VGPRs 4-7 hold
// K=base+16..base+23 (base already includes 8*half). Two aligned 16B runs.
static __device__ __forceinline__ v16bf fragFromBF(const unsigned short* p) {
  BFrag u;
  u.q[0] = *reinterpret_cast<const uint4*>(p);
  u.q[1] = *reinterpret_cast<const uint4*>(p + 16);
  return u.v;
}

extern "C" __global__ __launch_bounds__(512)
void pf_persistent(const float* __restrict__ obs,
                   const float* __restrict__ W0, const float* __restrict__ U0,
                   const float* __restrict__ b0,
                   const float* __restrict__ W1, const float* __restrict__ U1,
                   const float* __restrict__ b1,
                   const float* __restrict__ Wm, const float* __restrict__ bm,
                   float* __restrict__ out, float* __restrict__ ws)
{
  // LDS (<=320KB/WGP): 80K+64K bf16 weights [col][k], 128K fused ping-pong h panel,
  // ~8K control. ~282KB total; 1 WG per WGP by design (latency-bound scan).
  __shared__ __align__(16) unsigned short WT0[GATES * KA];
  __shared__ __align__(16) unsigned short WT1[GATES * KB];
  __shared__ __align__(16) unsigned short panels[2 * PANSZ];   // [rb]=h1old, [wb]=stage/h1new
  __shared__ __align__(16) unsigned short obsbf[DOBS];
  __shared__ float b0s[GATES], b1s[GATES], wms[DOBS + DS], obsf[DOBS];
  __shared__ float wts[NP], cum[NP], wpart[16];
  __shared__ unsigned short idxs[NP];

  const int b    = blockIdx.x;
  const int tid  = threadIdx.x;
  const int wave = tid >> 5;
  const int lane = tid & 31;
  const int half = lane >> 4;
  const int lrow = lane & 15;

  // ---- stage weights to LDS once: bf16, column-major [col][k]; k-major loop
  // (k = i>>8, n = i&255) avoids magic-number division ----
  for (int i = tid; i < GATES * KA; i += NP) {
    int k = i >> 8, n = i & (GATES - 1);
    float v = (k < 96) ? W0[k * GATES + n] : U0[(k - 96) * GATES + n];
    WT0[n * KA + k] = f2bf(v);
  }
  for (int i = tid; i < GATES * KB; i += NP) {
    int k = i >> 8, n = i & (GATES - 1);
    float v = (k < 64) ? W1[k * GATES + n] : U1[(k - 64) * GATES + n];
    WT1[n * KB + k] = f2bf(v);
  }
  if (tid < GATES) { b0s[tid] = b0[tid]; b1s[tid] = b1[tid]; }
  if (tid < DOBS + DS) wms[tid] = Wm[tid];
  wts[tid] = 1.0f / NP;
  const float bmv = bm[0];

  // workspace: c-states f32 ping-pong (recurrence precision), h0 bf16 ping-pong.
  // h1 never leaves LDS (panel ping-pong). Working set ~10MB -> resident in L2.
  const size_t SZ = (size_t)NB * NP * DS;
  unsigned short* hbase = reinterpret_cast<unsigned short*>(ws + 4 * SZ);

  __syncthreads();

  for (int t = 0; t < NT; ++t) {
    const float* obsrow = obs + ((size_t)b * NT + t) * DOBS;
    const int rb = t & 1, wb = rb ^ 1;
    const float* c0o = ws + (size_t)(rb * 2 + 0) * SZ;
    float*       c0n = ws + (size_t)(wb * 2 + 0) * SZ;
    const float* c1o = ws + (size_t)(rb * 2 + 1) * SZ;
    float*       c1n = ws + (size_t)(wb * 2 + 1) * SZ;
    const unsigned short* h0o = hbase + (size_t)rb * SZ;
    unsigned short*       h0n = hbase + (size_t)wb * SZ;
    const unsigned short* h1pan = panels + rb * PANSZ;
    unsigned short*       wpan  = panels + wb * PANSZ;

    // ---- stage obs row (f32 + bf16) once per step ----
    if (tid < DOBS) { float v = obsrow[tid]; obsf[tid] = v; obsbf[tid] = f2bf(v); }
    __syncthreads();

    // ---- resample: wave32 shuffle max/scan + 16-partial combine + inv-CDF ----
    if (t > 0) {
      const float w = wts[tid];
      float m = w;
#pragma unroll
      for (int off = 16; off >= 1; off >>= 1) m = fmaxf(m, __shfl_xor(m, off, 32));
      if (lane == 0) wpart[wave] = m;
      __syncthreads();
      if (tid < 16) {
        float v = wpart[tid];
#pragma unroll
        for (int off = 8; off >= 1; off >>= 1) v = fmaxf(v, __shfl_xor(v, off, 16));
        if (tid == 0) wpart[0] = v;
      }
      __syncthreads();
      const float wmax = wpart[0];
      __syncthreads();                       // wpart reused below
      float run = __expf(w - wmax);
#pragma unroll
      for (int off = 1; off < 32; off <<= 1) {
        float y = __shfl_up(run, off, 32);
        if (lane >= off) run += y;
      }
      if (lane == 31) wpart[wave] = run;
      __syncthreads();
      if (tid < 16) {
        float v = wpart[tid];
#pragma unroll
        for (int off = 1; off < 16; off <<= 1) {
          float y = __shfl_up(v, off, 16);
          if (tid >= off) v += y;
        }
        wpart[tid] = v;
      }
      __syncthreads();
      const float base = (wave == 0) ? 0.0f : wpart[wave - 1];
      cum[tid] = base + run;
      __syncthreads();
      const float tot = wpart[15];
      const float u = u01(hashu(((unsigned)((b * NP + tid) * NT + t)) ^ 0xDEADBEEFu)) * tot;
      int pos = 0;
#pragma unroll
      for (int off = 256; off >= 1; off >>= 1) {
        int np2 = pos + off;
        if (np2 <= NP - 1 && cum[np2 - 1] < u) pos = np2;
      }
      idxs[tid] = (unsigned short)pos;
      __syncthreads();
    }

    // ---- each wave owns two 16-particle M-tiles ----
    for (int mi = 0; mi < 2; ++mi) {
      const int mt = wave * 2 + mi;
      const int pg = mt * 16 + lrow;            // this lane's A-row particle
      const int g  = t ? (int)idxs[pg] : pg;    // resample gather folded into loads

      // ===== layer 0: z = [noise|obs|h0_gathered] @ [W0;U0] + b0 =====
      const unsigned seedb = (unsigned)((b * NP + pg) * NT + t) * 32u;
      v16bf A0[5];
#pragma unroll
      for (int kt = 0; kt < 2; ++kt) {          // k 0..63: noise, generated in pairs
        BFrag u;
#pragma unroll
        for (int e = 0; e < 16; e += 2) {       // pairs stay inside 8-wide K runs
          int d = 32 * kt + 8 * half + ((e < 8) ? e : e + 8);
          float n1, n2;
          nrm2(seedb + (unsigned)(d >> 1), &n1, &n2);
          u.s[e]     = f2bf(n1);
          u.s[e + 1] = f2bf(n2);
        }
        A0[kt] = u.v;
      }
      A0[2] = fragFromBF(obsbf + 8 * half);     // k 64..95: obs broadcast (LDS bf16)
      if (t) {                                  // k 96..159: gathered h0 (global bf16)
        const unsigned short* h0row = h0o + ((size_t)(b * NP + g)) * DS;
        A0[3] = fragFromBF(h0row + 8 * half);
        A0[4] = fragFromBF(h0row + 32 + 8 * half);
      } else { A0[3] = fragZero(); A0[4] = fragZero(); }

#pragma unroll
      for (int dt = 0; dt < 4; ++dt) {          // 16-wide slice of state dim
        v8f acc[4];
#pragma unroll
        for (int gi = 0; gi < 4; ++gi) {        // i,f,g,o gate tiles
          const int col = gi * 64 + dt * 16 + lrow;
          const float bb = b0s[col];
#pragma unroll
          for (int v = 0; v < 8; ++v) acc[gi][v] = bb;
#pragma unroll
          for (int kt = 0; kt < 5; ++kt) {
            v16bf Bf = fragFromBF(WT0 + col * KA + 32 * kt + 8 * half);
            acc[gi] = __builtin_amdgcn_wmma_f32_16x16x32_bf16(
                false, A0[kt], false, Bf, (short)0, acc[gi], false, false);
          }
        }
        const int dcol = dt * 16 + lrow;
#pragma unroll
        for (int v = 0; v < 8; ++v) {           // D layout: VGPR v -> row v+8*half
          const int pp = mt * 16 + v + 8 * half;
          const int gv = t ? (int)idxs[pp] : pp;
          const float c0 = t ? c0o[((size_t)(b * NP + gv)) * DS + dcol] : 0.0f;
          const float cn = sigm(acc[1][v]) * c0 + sigm(acc[0][v]) * tanh_f(acc[2][v]);
          const float hn = sigm(acc[3][v]) * tanh_f(cn);
          const unsigned short hb = f2bf(hn);
          const size_t gidx = ((size_t)(b * NP + pp)) * DS + dcol;
          c0n[gidx] = cn;
          h0n[gidx] = hb;                       // bf16 state: raw store, no cvt on load
          wpan[pp * DS + dcol] = hb;            // D->A transpose staging (LDS)
        }
      }

      // ===== layer 1: z = [h0new | h1old_gathered] @ [W1;U1] + b1 =====
      v16bf A1[4];
      A1[0] = fragFromBF(wpan + pg * DS + 8 * half);       // own-wave DS in-order
      A1[1] = fragFromBF(wpan + pg * DS + 32 + 8 * half);
      if (t) {                                  // h1old gathered from LDS panel
        const unsigned short* h1row = h1pan + g * DS;
        A1[2] = fragFromBF(h1row + 8 * half);
        A1[3] = fragFromBF(h1row + 32 + 8 * half);
      } else { A1[2] = fragZero(); A1[3] = fragZero(); }

#pragma unroll
      for (int dt = 0; dt < 4; ++dt) {
        v8f acc[4];
#pragma unroll
        for (int gi = 0; gi < 4; ++gi) {
          const int col = gi * 64 + dt * 16 + lrow;
          const float bb = b1s[col];
#pragma unroll
          for (int v = 0; v < 8; ++v) acc[gi][v] = bb;
#pragma unroll
          for (int kt = 0; kt < 4; ++kt) {
            v16bf Bf = fragFromBF(WT1 + col * KB + 32 * kt + 8 * half);
            acc[gi] = __builtin_amdgcn_wmma_f32_16x16x32_bf16(
                false, A1[kt], false, Bf, (short)0, acc[gi], false, false);
          }
        }
        const int dcol = dt * 16 + lrow;
#pragma unroll
        for (int v = 0; v < 8; ++v) {
          const int pp = mt * 16 + v + 8 * half;
          const int gv = t ? (int)idxs[pp] : pp;
          const float c1 = t ? c1o[((size_t)(b * NP + gv)) * DS + dcol] : 0.0f;
          const float cn = sigm(acc[1][v]) * c1 + sigm(acc[0][v]) * tanh_f(acc[2][v]);
          const float hn = sigm(acc[3][v]) * tanh_f(cn);
          const size_t gidx = ((size_t)(b * NP + pp)) * DS + dcol;
          c1n[gidx] = cn;
          wpan[pp * DS + dcol] = f2bf(hn);      // overwrite staging with h1new
        }
      }
    }

    __threadfence_block();
    __syncthreads();

    // ---- measurement: w[n] = [obs | h1[n]] @ Wm + bm ----
    {
      float w = bmv;
#pragma unroll
      for (int j = 0; j < DOBS; ++j) w += obsf[j] * wms[j];
      const unsigned short* hrow = wpan + tid * DS;
      float s = 0.0f;
#pragma unroll
      for (int d = 0; d < DS; ++d) s += bf2f(hrow[d]) * wms[DOBS + d];
      w += s;
      wts[tid] = w;
      if (t == NT - 1) {                        // out[b][n] = concat(h1, w)
        float* orow = out + ((size_t)(b * NP + tid)) * (DS + 1);
#pragma unroll
        for (int d = 0; d < DS; ++d) orow[d] = bf2f(hrow[d]);
        orow[DS] = w;
      }
    }
    __syncthreads();
  }
}

extern "C" void kernel_launch(void* const* d_in, const int* in_sizes, int n_in,
                              void* d_out, int out_size, void* d_ws, size_t ws_size,
                              hipStream_t stream) {
  const float* obs = (const float*)d_in[0];
  const float* W0  = (const float*)d_in[1];
  const float* U0  = (const float*)d_in[2];
  const float* b0  = (const float*)d_in[3];
  const float* W1  = (const float*)d_in[4];
  const float* U1  = (const float*)d_in[5];
  const float* b1  = (const float*)d_in[6];
  const float* Wm  = (const float*)d_in[7];
  const float* bm  = (const float*)d_in[8];
  (void)in_sizes; (void)n_in; (void)out_size; (void)ws_size;
  pf_persistent<<<NB, NP, 0, stream>>>(obs, W0, U0, b0, W1, U1, b1, Wm, bm,
                                       (float*)d_out, (float*)d_ws);
}
